// MambaBlock_12163347382741
// MI455X (gfx1250) — compile-verified, matrix operand order checked by reading, not run on hardware
//
#include <hip/hip_runtime.h>
#include <hip/hip_bf16.h>

// Mamba block for MI455X (gfx1250, wave32, WMMA).
// All three big GEMMs run on v_wmma_f32_16x16x32_bf16 with operands
// pre-converted to bf16 (no conversion VALU in the WMMA hot loop; fragments
// load as raw global_load_b128). Selective scan keeps h[16] per-channel in
// VGPRs (one thread per (b,d), 16-way ILP on the v_exp_f32 chain).

#define BB   2
#define LSEQ 2048
#define DM   768
#define DI   1536          // d_inner
#define DXZ  (2*DI)        // 3072

typedef __attribute__((ext_vector_type(16))) __bf16 v16bf;
typedef __attribute__((ext_vector_type(8)))  float  v8f;

__device__ __forceinline__ unsigned short f32_to_bf16_rne(float f) {
  union { float f; unsigned int u; } c; c.f = f;
  unsigned int u = c.u + 0x7FFFu + ((c.u >> 16) & 1u);
  return (unsigned short)(u >> 16);
}

// WMMA 16x16x32 bf16 fragment, loaded straight from a K-major bf16 matrix.
// ISA layout (CDNA5 ISA): lane L holds row/col (L&15); lanes 0-15 carry
// K = {0..7, 16..23}, lanes 16-31 K = {8..15, 24..31}. Each 8-element run is
// 16 contiguous bytes -> two global_load_b128 per fragment.
__device__ __forceinline__ v16bf load_frag(const unsigned short* __restrict__ p) {
  union { v16bf v; uint4 u[2]; } r;
  r.u[0] = *(const uint4*)(p);        // K run at +0
  r.u[1] = *(const uint4*)(p + 16);   // K run at +16
  return r.v;
}

// Pack fp32 -> bf16 (RNE), 8 elements per thread, b128 in / b128 out.
__global__ void pack_bf16_kernel(const float* __restrict__ in,
                                 unsigned short* __restrict__ out, long n)
{
  long i = ((long)blockIdx.x * blockDim.x + threadIdx.x) * 8;
  if (i >= n) return;
  float4 a = *(const float4*)(in + i);
  float4 b = *(const float4*)(in + i + 4);
  union { unsigned short s[8]; uint4 u; } r;
  r.s[0]=f32_to_bf16_rne(a.x); r.s[1]=f32_to_bf16_rne(a.y);
  r.s[2]=f32_to_bf16_rne(a.z); r.s[3]=f32_to_bf16_rne(a.w);
  r.s[4]=f32_to_bf16_rne(b.x); r.s[5]=f32_to_bf16_rne(b.y);
  r.s[6]=f32_to_bf16_rne(b.z); r.s[7]=f32_to_bf16_rne(b.w);
  *(uint4*)(out + i) = r.u;
}

// C(MxN) = A(MxK) * B(NxK)^T, bf16 operands, f32 accumulate/output.
// Block = 256 threads = 8 waves arranged 2(M) x 4(N); each wave owns a
// 16x64 tile: one A fragment reused across 4 B fragments -> 4 WMMAs per
// 10 b128 loads per 32-K step. Requires M%32==0, N%256==0, K%32==0.
// SOFTPLUS fuses val = softplus(val + bias[n]) (delta path).
template<bool SOFTPLUS>
__global__ __launch_bounds__(256) void wmma_gemm_bf16(
    const unsigned short* __restrict__ A, const unsigned short* __restrict__ Bw,
    float* __restrict__ C, const float* __restrict__ bias,
    int K, long lda, long ldb, long ldc)
{
  const int wave = threadIdx.x >> 5;
  const int lane = threadIdx.x & 31;
  const int wm = wave >> 2;           // 0..1
  const int wn = wave & 3;            // 0..3
  const long m0 = ((long)blockIdx.y * 2 + wm) * 16;
  const long n0 = ((long)blockIdx.x * 4 + wn) * 64;
  const int r   = lane & 15;
  const int kb  = (lane >> 4) << 3;   // 0 or 8

  const unsigned short* pa = A  + (m0 + r) * lda + kb;
  const unsigned short* pb = Bw + (n0 + r) * ldb + kb;
  const long bstep = 16 * ldb;        // next 16-column B tile

  v8f acc0 = {}, acc1 = {}, acc2 = {}, acc3 = {};
  for (int k0 = 0; k0 < K; k0 += 32) {
    v16bf a  = load_frag(pa + k0);
    v16bf b0 = load_frag(pb + k0);
    v16bf b1 = load_frag(pb + bstep     + k0);
    v16bf b2 = load_frag(pb + 2*bstep   + k0);
    v16bf b3 = load_frag(pb + 3*bstep   + k0);
    acc0 = __builtin_amdgcn_wmma_f32_16x16x32_bf16(false, a, false, b0, (short)0, acc0, false, false);
    acc1 = __builtin_amdgcn_wmma_f32_16x16x32_bf16(false, a, false, b1, (short)0, acc1, false, false);
    acc2 = __builtin_amdgcn_wmma_f32_16x16x32_bf16(false, a, false, b2, (short)0, acc2, false, false);
    acc3 = __builtin_amdgcn_wmma_f32_16x16x32_bf16(false, a, false, b3, (short)0, acc3, false, false);
  }

  // D layout: lanes 0-15 -> rows m0+v, lanes 16-31 -> rows m0+8+v; col = (lane&15)
  const long mrow = m0 + ((lane >> 4) << 3);
  v8f accs[4] = {acc0, acc1, acc2, acc3};
  #pragma unroll
  for (int t = 0; t < 4; ++t) {
    const long col = n0 + t * 16 + r;
    float bv = SOFTPLUS ? bias[col] : 0.f;
    #pragma unroll
    for (int v = 0; v < 8; ++v) {
      float val = accs[t][v];
      if (SOFTPLUS) {
        val += bv;
        val = (val > 20.f) ? val : log1pf(__expf(val));   // stable softplus
      }
      C[(mrow + v) * ldc + col] = val;
    }
  }
}

// Depthwise conv1d (k=3, pad=1) + bias + SiLU over the xi half of xz.
// Writes fp32 (for scan/BC) and bf16 (GEMM2 A-operand).
__global__ void conv1d_silu_kernel(const float* __restrict__ xz,
                                   const float* __restrict__ w,
                                   const float* __restrict__ bias,
                                   float* __restrict__ xi_act,
                                   unsigned short* __restrict__ xi_bf)
{
  long idx = (long)blockIdx.x * blockDim.x + threadIdx.x;
  const long total = (long)BB * LSEQ * DI;
  if (idx >= total) return;
  int  d  = (int)(idx % DI);
  long bl = idx / DI;
  int  l  = (int)(bl % LSEQ);
  long b  = bl / LSEQ;
  const float* xp = xz + b * LSEQ * (long)DXZ;
  float w0 = w[d*3+0], w1 = w[d*3+1], w2 = w[d*3+2];
  float s = bias[d];
  if (l > 0)        s += xp[(long)(l-1)*DXZ + d] * w0;
  s += xp[(long)l*DXZ + d] * w1;
  if (l < LSEQ-1)   s += xp[(long)(l+1)*DXZ + d] * w2;
  float act = s / (1.f + __expf(-s));            // silu
  xi_act[idx] = act;
  xi_bf[idx]  = f32_to_bf16_rne(act);
}

// BC[b,l,0..31] = xi_act[b,l,:] @ W_x[n,:]^T ; one wave per (b,l), lane = n.
__global__ __launch_bounds__(256) void bc_kernel(const float* __restrict__ xi_act,
                                                 const float* __restrict__ Wx,
                                                 float* __restrict__ BC)
{
  int wg   = blockIdx.x * 8 + (threadIdx.x >> 5);   // (b*L + l)
  int lane = threadIdx.x & 31;                      // n in [0,32)
  if (wg >= BB * LSEQ) return;
  const float* xp = xi_act + (long)wg * DI;
  const float* wp = Wx + (long)lane * DI;
  float s = 0.f;
  for (int k = 0; k < DI; k += 4) {
    float4 xv = *(const float4*)(xp + k);
    float4 wv = *(const float4*)(wp + k);
    s += xv.x*wv.x + xv.y*wv.y + xv.z*wv.z + xv.w*wv.w;
  }
  BC[(long)wg * 32 + lane] = s;
}

// Selective scan fused with "+ x*D" and the silu(z) gate; emits bf16 y
// (GEMM3 A-operand) directly. One thread per (b,d); h[16] in VGPRs.
__global__ void scan_kernel(const float* __restrict__ xz,
                            const float* __restrict__ xi_act,
                            const float* __restrict__ BC,
                            const float* __restrict__ A_log,
                            const float* __restrict__ Dp,
                            unsigned short* __restrict__ y_bf)
{
  int idx = blockIdx.x * blockDim.x + threadIdx.x;
  if (idx >= BB * DI) return;
  int d = idx % DI;
  int b = idx / DI;

  float Ac[16];
  #pragma unroll
  for (int n = 0; n < 16; ++n) Ac[n] = -__expf(A_log[(long)d * 16 + n]);
  float Dv = Dp[d];
  float h[16];
  #pragma unroll
  for (int n = 0; n < 16; ++n) h[n] = 0.f;

  const float* dz = xz     + (long)b * LSEQ * DXZ;
  const float* xp = xi_act + (long)b * LSEQ * DI;
  const float* bc = BC     + (long)b * LSEQ * 32;
  unsigned short* yp = y_bf + (long)b * LSEQ * DI;

  for (int l = 0; l < LSEQ; ++l) {
    float dl = dz[(long)l * DXZ + d];          // softplus'ed delta
    float zv = dz[(long)l * DXZ + DI + d];     // gate input
    float xv = xp[(long)l * DI + d];           // xi (post conv+silu)
    const float* bcl = bc + l * 32;            // B[0..15], C[16..31] (uniform)
    float y  = xv * Dv;
    float dx = dl * xv;
    #pragma unroll
    for (int n = 0; n < 16; ++n) {
      float dA = __expf(dl * Ac[n]);           // v_exp_f32 chain, 16-way ILP
      h[n] = dA * h[n] + bcl[n] * dx;
      y   += h[n] * bcl[16 + n];
    }
    float g = zv / (1.f + __expf(-zv));        // silu(z)
    yp[(long)l * DI + d] = f32_to_bf16_rne(y * g);
  }
}

extern "C" void kernel_launch(void* const* d_in, const int* in_sizes, int n_in,
                              void* d_out, int out_size, void* d_ws, size_t ws_size,
                              hipStream_t stream)
{
  const float* x      = (const float*)d_in[0];
  const float* W_in   = (const float*)d_in[1];
  const float* conv_w = (const float*)d_in[2];
  const float* conv_b = (const float*)d_in[3];
  const float* W_x    = (const float*)d_in[4];
  const float* W_dt   = (const float*)d_in[5];
  const float* b_dt   = (const float*)d_in[6];
  const float* A_log  = (const float*)d_in[7];
  const float* D_par  = (const float*)d_in[8];
  const float* W_out  = (const float*)d_in[9];
  float* out = (float*)d_out;

  const long M = (long)BB * LSEQ;  // 4096 rows

  // Workspace layout (~103 MB):
  //   xz fp32 [M x 3072]   (GEMM1 out; delta overwrites [:,0:DI], z in [:,DI:2DI])
  //   xi_act fp32 [M x DI] (conv+silu, fp32 for scan/BC)
  //   BC fp32 [M x 32]
  //   xi_bf  bf16 [M x DI]
  //   Wdt_bf bf16 [DI x DI], Wout_bf bf16 [DM x DI]
  //   union region: {x_bf [M x DM] + Win_bf [DXZ x DM]} reused later as y_bf [M x DI]
  char* wsp = (char*)d_ws;
  float* xz     = (float*)wsp;                 wsp += (size_t)M * DXZ * 4;
  float* xi_act = (float*)wsp;                 wsp += (size_t)M * DI * 4;
  float* BC     = (float*)wsp;                 wsp += (size_t)M * 32 * 4;
  unsigned short* xi_bf   = (unsigned short*)wsp; wsp += (size_t)M * DI * 2;
  unsigned short* Wdt_bf  = (unsigned short*)wsp; wsp += (size_t)DI * DI * 2;
  unsigned short* Wout_bf = (unsigned short*)wsp; wsp += (size_t)DM * DI * 2;
  unsigned short* x_bf    = (unsigned short*)wsp;                   // M*DM
  unsigned short* Win_bf  = x_bf + (size_t)M * DM;                  // DXZ*DM
  unsigned short* y_bf    = x_bf;                                   // M*DI (after GEMM1)

  // 0) one-time bf16 packs (x + the three WMMA weight matrices)
  {
    long n;
    n = M * DM;            pack_bf16_kernel<<<(int)(n/8/256), 256, 0, stream>>>(x, x_bf, n);
    n = (long)DXZ * DM;    pack_bf16_kernel<<<(int)(n/8/256), 256, 0, stream>>>(W_in, Win_bf, n);
    n = (long)DI * DI;     pack_bf16_kernel<<<(int)(n/8/256), 256, 0, stream>>>(W_dt, Wdt_bf, n);
    n = (long)DM * DI;     pack_bf16_kernel<<<(int)(n/8/256), 256, 0, stream>>>(W_out, Wout_bf, n);
  }

  // 1) xz = x @ W_in^T          (M=4096, N=3072, K=768)
  wmma_gemm_bf16<false><<<dim3(DXZ/256, M/32), 256, 0, stream>>>(
      x_bf, Win_bf, xz, nullptr, DM, DM, DM, DXZ);

  // 2) xi = silu(conv1d(xi) + b) -> fp32 + bf16
  {
    long total = M * (long)DI;
    conv1d_silu_kernel<<<(int)((total + 255) / 256), 256, 0, stream>>>(
        xz, conv_w, conv_b, xi_act, xi_bf);
  }

  // 3) BC = xi @ W_x^T           (N=32 -> wave-reduction kernel)
  bc_kernel<<<(int)((M + 7) / 8), 256, 0, stream>>>(xi_act, W_x, BC);

  // 4) delta = softplus(xi @ W_dt^T + b_dt), stored into xz[:, 0:DI]
  wmma_gemm_bf16<true><<<dim3(DI/256, M/32), 256, 0, stream>>>(
      xi_bf, Wdt_bf, xz, b_dt, DI, DI, DI, DXZ);

  // 5) selective scan + x*D + silu(z) gate -> bf16 y (overwrites x_bf/Win_bf region)
  scan_kernel<<<(BB * DI + 127) / 128, 128, 0, stream>>>(
      xz, xi_act, BC, A_log, D_par, y_bf);

  // 6) out = y @ W_out^T         (M=4096, N=768, K=1536)
  wmma_gemm_bf16<false><<<dim3(DM/256, M/32), 256, 0, stream>>>(
      y_bf, Wout_bf, out, nullptr, DI, DI, DI, DM);
}
